// P2A2Module_4509715661187
// MI455X (gfx1250) — compile-verified, matrix operand order checked by reading, not run on hardware
//
#include <hip/hip_runtime.h>
#include <math.h>

// ---------------------------------------------------------------------------
// Shapes (compile-time constants from the reference)
// ---------------------------------------------------------------------------
#define BSZ   32
#define DIM   128
#define KD    16
#define NH    8
#define DV    64          // 4*KD
#define DH    512         // DV*NH
#define NHKD  128         // KD*NH
#define PE_L  16
#define HH    64
#define WW    64
#define HW    4096        // HH*WW
#define BEPS  1e-5f

#define POSB  128         // positions per block (8 waves x 16)
#define MB    64          // output channels per block (4 WMMA M-tiles / wave)

typedef __attribute__((ext_vector_type(16))) _Float16 v16h;
typedef __attribute__((ext_vector_type(8)))  _Float16 v8h;
typedef __attribute__((ext_vector_type(8)))  float    v8f;
typedef __attribute__((ext_vector_type(2)))  _Float16 h2f;

// ---------------------------------------------------------------------------
// B-slab staging: 32(K) x 128(P) f16 -> LDS, fragment-major
// (half index = group*512 + n*32 + k).  Each of 256 threads handles a k-PAIR
// over 8 positions: 2x 16B global loads, 8 packed ds_store_b32, zero VALU cvt.
// ---------------------------------------------------------------------------
__device__ __forceinline__ void stage_slab_h(
    _Float16* lds, const _Float16* __restrict__ xr0, const _Float16* __restrict__ xr1,
    int tid, int pb, int Npos, bool full)
{
    const int kk  = (tid & 15) * 2;
    const int sub = tid >> 4;                       // 16 subgroups of 8 positions
    h2f* dst = (h2f*)lds + (sub >> 1) * 256 + ((sub & 1) * 8) * 16 + (kk >> 1);
    if (full) {
        v8h a = *(const v8h*)xr0;
        v8h b = *(const v8h*)xr1;
#pragma unroll
        for (int j = 0; j < 8; ++j) {
            h2f hv = {a[j], b[j]};
            dst[j * 16] = hv;
        }
    } else {
#pragma unroll
        for (int j = 0; j < 8; ++j) {
            bool ok = (pb + j) < Npos;
            h2f hv = {ok ? xr0[j] : (_Float16)0.0f, ok ? xr1[j] : (_Float16)0.0f};
            dst[j * 16] = hv;
        }
    }
}

// ---------------------------------------------------------------------------
// Generic batched GEMM:  Y[b, m, p] = epilogue( sum_k W[m,k] * X[b,k,p] )
// A and B operands are pre-converted f16.  Block: 64(M) x 128(P); each wave
// runs 4 M-tiles sharing one B fragment -> 32 WMMAs per staged 8KB slab.
// Epilogue (all optional): +rowAdd/+colAdd broadcast, BN, PReLU, +res, *mul;
// writes f32 Y and/or an f16 copy Yh for downstream GEMM consumers.
// ---------------------------------------------------------------------------
__global__ __launch_bounds__(256) void wmma_gemm_kernel(
    const _Float16* __restrict__ Wh,  // [Cout, Cin] f16
    const _Float16* __restrict__ Xh,  // [B, Cin, Npos] f16
    float* __restrict__ Y,            // [B, Cout, Npos] f32 (optional)
    _Float16* __restrict__ Yh,        // [B, Cout, Npos] f16 (optional)
    int Cout, int Cin, int Npos,
    const float* __restrict__ bn_g, const float* __restrict__ bn_b,
    const float* __restrict__ bn_m, const float* __restrict__ bn_v,
    const float* __restrict__ prelu_a,
    const float* __restrict__ res,    // [B, Cout, Npos]
    const float* __restrict__ mul,    // [B, Cout, Npos]
    const float* __restrict__ rowAdd, // [B, Cout, Npos/Wd]
    const float* __restrict__ colAdd, // [B, Cout, Wd]
    int Wd)
{
    __shared__ _Float16 lds[8 * 512];
    const int tid  = threadIdx.x;
    const int lane = tid & 31;
    const int wave = tid >> 5;
    const int b    = blockIdx.z;
    const int m0   = blockIdx.y * MB;
    const int pos0 = blockIdx.x * POSB;
    const int p0   = pos0 + wave * 16;
    const bool full = (pos0 + POSB) <= Npos;       // block-uniform

    const int lkk = (tid & 15) * 2;
    const int lpb = pos0 + (tid >> 4) * 8;
    const int lpc = full ? lpb : min(lpb, Npos - 8 < 0 ? 0 : Npos - 8);

    const size_t xb = (size_t)b * Cin * Npos;
    v8f acc0 = {}, acc1 = {}, acc2 = {}, acc3 = {};

    const int arow = lane & 15;
    const int acol = (lane >> 4) * 16;
    const _Float16* bsrc = &lds[wave * 512 + (lane & 15) * 32 + (lane >> 4) * 16];

    for (int k0 = 0; k0 < Cin; k0 += 32) {
        const _Float16* xr0 = Xh + xb + (size_t)(k0 + lkk) * Npos + lpc;
        stage_slab_h(lds, xr0, xr0 + Npos, tid, lpb, Npos, full);
        if (k0 + 32 < Cin)
            __builtin_prefetch(Xh + xb + (size_t)(k0 + 32 + lkk) * Npos + lpc, 0, 1);
        __syncthreads();

        v16h bfrag = *(const v16h*)bsrc;
        const _Float16* wr = Wh + (size_t)(m0 + arow) * Cin + k0 + acol;
        v16h a0 = *(const v16h*)(wr);
        v16h a1 = *(const v16h*)(wr + 16 * (size_t)Cin);
        v16h a2 = *(const v16h*)(wr + 32 * (size_t)Cin);
        v16h a3 = *(const v16h*)(wr + 48 * (size_t)Cin);
        acc0 = __builtin_amdgcn_wmma_f32_16x16x32_f16(false, a0, false, bfrag, (short)0, acc0, false, false);
        acc1 = __builtin_amdgcn_wmma_f32_16x16x32_f16(false, a1, false, bfrag, (short)0, acc1, false, false);
        acc2 = __builtin_amdgcn_wmma_f32_16x16x32_f16(false, a2, false, bfrag, (short)0, acc2, false, false);
        acc3 = __builtin_amdgcn_wmma_f32_16x16x32_f16(false, a3, false, bfrag, (short)0, acc3, false, false);
        __syncthreads();
    }

    // ---- epilogue ----
    const int n   = lane & 15;
    const int mhi = (lane >> 4) * 8;
    const int pos = p0 + n;
    v8f accs[4] = {acc0, acc1, acc2, acc3};
#pragma unroll
    for (int mt = 0; mt < 4; ++mt) {
#pragma unroll
        for (int r = 0; r < 8; ++r) {
            int m = m0 + mt * 16 + mhi + r;
            float v = accs[mt][r];
            if (rowAdd) {
                int h = pos / Wd, w = pos % Wd;
                v += rowAdd[((size_t)b * Cout + m) * (Npos / Wd) + h]
                   + colAdd[((size_t)b * Cout + m) * Wd + w];
            }
            if (bn_g) {
                float sc = bn_g[m] * rsqrtf(bn_v[m] + BEPS);
                v = (v - bn_m[m]) * sc + bn_b[m];
            }
            if (prelu_a) v = (v >= 0.0f) ? v : prelu_a[m] * v;
            size_t oi = ((size_t)b * Cout + m) * Npos + pos;
            if (res) v += res[oi];
            if (mul) v *= mul[oi];
            if (full || pos < Npos) {
                if (Y)  Y[oi]  = v;
                if (Yh) Yh[oi] = (_Float16)v;
            }
        }
    }
}

// ---------------------------------------------------------------------------
// Implicit-GEMM 3x3 conv (stride 1, pad 1) + BN + residual:
//   qkv_details = BN(conv3x3(rh, de_w2)) + r
// Weights pre-transposed to [tap][co][ci] f16 -> contiguous v16h A loads.
// Input rh is the f16 copy of r; residual read from f32 r.
// ---------------------------------------------------------------------------
__global__ __launch_bounds__(256) void wmma_conv3x3_kernel(
    const _Float16* __restrict__ W3t, // [9,128,128] f16, [tap][co][ci]
    const _Float16* __restrict__ Xh,  // [B,128,64,64] f16
    float* __restrict__ Y,
    const float* __restrict__ bn_g, const float* __restrict__ bn_b,
    const float* __restrict__ bn_m, const float* __restrict__ bn_v,
    const float* __restrict__ res)
{
    __shared__ _Float16 lds[8 * 512];
    const int tid  = threadIdx.x;
    const int lane = tid & 31;
    const int wave = tid >> 5;
    const int b    = blockIdx.z;
    const int m0   = blockIdx.y * MB;
    const int pos0 = blockIdx.x * POSB;
    const int p0   = pos0 + wave * 16;

    const int lkk = (tid & 15) * 2;
    const int sub = tid >> 4;
    const int lpb = pos0 + sub * 8;                // 8 positions, same image row
    const int lh  = lpb >> 6;
    const int lw  = lpb & 63;

    v8f acc0 = {}, acc1 = {}, acc2 = {}, acc3 = {};
    const int arow = lane & 15;
    const int acol = (lane >> 4) * 16;
    const _Float16* bsrc = &lds[wave * 512 + (lane & 15) * 32 + (lane >> 4) * 16];

    for (int ky = 0; ky < 3; ++ky)
    for (int kx = 0; kx < 3; ++kx)
    for (int k0 = 0; k0 < DIM; k0 += 32) {
        // ---- stage shifted slab (k-pair, 8 positions) with halo zeros ----
        {
            const _Float16* xc0 = Xh + ((size_t)b * DIM + k0 + lkk) * HW;
            const _Float16* xc1 = xc0 + HW;
            int hh = lh + ky - 1;
            bool hok = (hh >= 0) && (hh < HH);
            h2f* dst = (h2f*)lds + (sub >> 1) * 256 + ((sub & 1) * 8) * 16 + (lkk >> 1);
#pragma unroll
            for (int j = 0; j < 8; ++j) {
                int ww = lw + j + kx - 1;
                bool ok = hok && (ww >= 0) && (ww < WW);
                int off = hh * WW + ww;
                h2f hv = {ok ? xc0[off] : (_Float16)0.0f, ok ? xc1[off] : (_Float16)0.0f};
                dst[j * 16] = hv;
            }
        }
        __syncthreads();

        v16h bfrag = *(const v16h*)bsrc;
        const _Float16* wr = W3t + (size_t)(ky * 3 + kx) * DIM * DIM
                                 + (size_t)(m0 + arow) * DIM + k0 + acol;
        v16h a0 = *(const v16h*)(wr);
        v16h a1 = *(const v16h*)(wr + 16 * DIM);
        v16h a2 = *(const v16h*)(wr + 32 * DIM);
        v16h a3 = *(const v16h*)(wr + 48 * DIM);
        acc0 = __builtin_amdgcn_wmma_f32_16x16x32_f16(false, a0, false, bfrag, (short)0, acc0, false, false);
        acc1 = __builtin_amdgcn_wmma_f32_16x16x32_f16(false, a1, false, bfrag, (short)0, acc1, false, false);
        acc2 = __builtin_amdgcn_wmma_f32_16x16x32_f16(false, a2, false, bfrag, (short)0, acc2, false, false);
        acc3 = __builtin_amdgcn_wmma_f32_16x16x32_f16(false, a3, false, bfrag, (short)0, acc3, false, false);
        __syncthreads();
    }

    const int n   = lane & 15;
    const int mhi = (lane >> 4) * 8;
    const int pos = p0 + n;
    v8f accs[4] = {acc0, acc1, acc2, acc3};
#pragma unroll
    for (int mt = 0; mt < 4; ++mt) {
#pragma unroll
        for (int r = 0; r < 8; ++r) {
            int m = m0 + mt * 16 + mhi + r;
            float sc = bn_g[m] * rsqrtf(bn_v[m] + BEPS);
            float v  = (accs[mt][r] - bn_m[m]) * sc + bn_b[m];
            size_t oi = ((size_t)b * DIM + m) * HW + pos;
            Y[oi] = v + res[oi];
        }
    }
}

// ---------------------------------------------------------------------------
// f32 -> f16 conversion (grid sized exactly to n/256 blocks)
// ---------------------------------------------------------------------------
__global__ void cvt_h_kernel(const float* __restrict__ in, _Float16* __restrict__ out)
{
    size_t i = (size_t)blockIdx.x * blockDim.x + threadIdx.x;
    out[i] = (_Float16)in[i];
}

// de_w2 [co][ci][ky][kx] f32 -> [tap][co][ci] f16
__global__ void cvt_w3_kernel(const float* __restrict__ w3, _Float16* __restrict__ out)
{
    size_t i = (size_t)blockIdx.x * blockDim.x + threadIdx.x;   // 9*128*128
    int tap = (int)(i >> 14);
    int co  = (int)((i >> 7) & 127);
    int ci  = (int)(i & 127);
    int ky = tap / 3, kx = tap % 3;
    out[i] = (_Float16)w3[(((size_t)co * DIM + ci) * 3 + ky) * 3 + kx];
}

// ---------------------------------------------------------------------------
// Separable average pooling (count_include_pad, zero borders).
// ---------------------------------------------------------------------------
__global__ void pool_h_kernel(const float* __restrict__ in, float* __restrict__ out, int k)
{
    size_t idx = (size_t)blockIdx.x * blockDim.x + threadIdx.x;
    int w = idx & 63, h = (idx >> 6) & 63;
    size_t base = idx - (h * WW + w);
    int pad = k >> 1;
    float s = 0.0f;
    for (int d = 0; d < k; ++d) {
        int ww = w + d - pad;
        if (ww >= 0 && ww < WW) s += in[base + h * WW + ww];
    }
    out[idx] = s;
}

__global__ void pool_v_kernel(const float* __restrict__ in, float* __restrict__ out, int k)
{
    size_t idx = (size_t)blockIdx.x * blockDim.x + threadIdx.x;
    int w = idx & 63, h = (idx >> 6) & 63;
    size_t base = idx - (h * WW + w);
    int pad = k >> 1;
    float s = 0.0f;
    for (int d = 0; d < k; ++d) {
        int hh = h + d - pad;
        if (hh >= 0 && hh < HH) s += in[base + hh * WW + w];
    }
    out[idx] = s * (1.0f / (float)(k * k));
}

// ---------------------------------------------------------------------------
// Depthwise 3x3 over cat(p1,p1,p3) (groups=128) + BN + PReLU, then * c1.
// Writes f32 xp (means) and f16 xph (final GEMM B operand).
// ---------------------------------------------------------------------------
__global__ void dwconv_kernel(
    const float* __restrict__ p1, const float* __restrict__ p3,
    const float* __restrict__ w2,               // [128,3,3,3]
    const float* __restrict__ bn_g, const float* __restrict__ bn_b,
    const float* __restrict__ bn_m, const float* __restrict__ bn_v,
    const float* __restrict__ a2,
    const float* __restrict__ c1,
    float* __restrict__ xp, _Float16* __restrict__ xph)
{
    size_t idx = (size_t)blockIdx.x * blockDim.x + threadIdx.x;
    int w = idx & 63, h = (idx >> 6) & 63;
    int c = (idx >> 12) & 127;
    int b = idx >> 19;
    float acc = 0.0f;
#pragma unroll
    for (int t = 0; t < 3; ++t) {
        int ch3 = 3 * c + t;
        const float* src = (ch3 < 256)
            ? (p1 + ((size_t)b * DIM + (ch3 & 127)) * HW)
            : (p3 + ((size_t)b * DIM + (ch3 - 256)) * HW);
#pragma unroll
        for (int ky = 0; ky < 3; ++ky) {
            int hh = h + ky - 1;
            if (hh < 0 || hh >= HH) continue;
#pragma unroll
            for (int kx = 0; kx < 3; ++kx) {
                int ww = w + kx - 1;
                if (ww < 0 || ww >= WW) continue;
                acc += w2[((c * 3 + t) * 3 + ky) * 3 + kx] * src[hh * WW + ww];
            }
        }
    }
    float sc = bn_g[c] * rsqrtf(bn_v[c] + BEPS);
    float v  = (acc - bn_m[c]) * sc + bn_b[c];
    v = (v >= 0.0f) ? v : a2[c] * v;
    v *= c1[idx];
    xp[idx]  = v;
    xph[idx] = (_Float16)v;
}

// ---------------------------------------------------------------------------
// Row/col means of x_pooled -> f16 (they only feed GEMMs).
// ---------------------------------------------------------------------------
__global__ void mean_kernel(const float* __restrict__ xp,
                            _Float16* __restrict__ xprh, _Float16* __restrict__ xpch)
{
    size_t idx = (size_t)blockIdx.x * blockDim.x + threadIdx.x;  // B*128*64
    int i  = idx & 63;
    size_t bc = idx >> 6;
    const float* base = xp + bc * HW;
    float sr = 0.0f, scol = 0.0f;
    for (int j = 0; j < 64; ++j) {
        sr   += base[i * WW + j];
        scol += base[j * WW + i];
    }
    xprh[idx] = (_Float16)(sr * (1.0f / 64.0f));
    xpch[idx] = (_Float16)(scol * (1.0f / 64.0f));
}

// ---------------------------------------------------------------------------
// Axial attention over one orientation (n = 64); output f16 (feeds GEMM).
// ---------------------------------------------------------------------------
__device__ __forceinline__ float pe_interp(const float* pe, int i)
{
    float src = ((float)i + 0.5f) * (16.0f / 64.0f) - 0.5f;
    src = fminf(fmaxf(src, 0.0f), 15.0f);
    int lo = (int)floorf(src);
    int hi = min(lo + 1, 15);
    float w = src - (float)lo;
    return pe[lo] * (1.0f - w) + pe[hi] * w;
}

__global__ __launch_bounds__(64) void axial_kernel(
    const float* __restrict__ qm,  // [B,128,64] f32
    const float* __restrict__ km,  // [B,128,64] f32
    const float* __restrict__ vm,  // [B,512,64] f32
    const float* __restrict__ peq, // [128,16]
    const float* __restrict__ pek, // [128,16]
    _Float16* __restrict__ outT)   // [B,512,64] f16
{
    __shared__ float qh[KD * 64];
    __shared__ float kh[KD * 64];
    __shared__ float vh[DV * 64];
    const int i  = threadIdx.x;
    const int nh = blockIdx.x & (NH - 1);
    const int b  = blockIdx.x >> 3;

    for (int kd = 0; kd < KD; ++kd) {
        int ch = nh * KD + kd;
        qh[kd * 64 + i] = qm[((size_t)b * NHKD + ch) * 64 + i] + pe_interp(peq + ch * PE_L, i);
        kh[kd * 64 + i] = km[((size_t)b * NHKD + ch) * 64 + i] + pe_interp(pek + ch * PE_L, i);
    }
    for (int d = 0; d < DV; ++d)
        vh[d * 64 + i] = vm[((size_t)b * DH + nh * DV + d) * 64 + i];
    __syncthreads();

    float s[64];
    float mx = -3.0e38f;
    for (int j = 0; j < 64; ++j) {
        float dot = 0.0f;
        for (int kd = 0; kd < KD; ++kd) dot += qh[kd * 64 + i] * kh[kd * 64 + j];
        s[j] = dot * 0.25f;
        mx = fmaxf(mx, s[j]);
    }
    float sum = 0.0f;
    for (int j = 0; j < 64; ++j) { s[j] = __expf(s[j] - mx); sum += s[j]; }
    float inv = 1.0f / sum;
    for (int d = 0; d < DV; ++d) {
        float o = 0.0f;
        for (int j = 0; j < 64; ++j) o += s[j] * vh[d * 64 + j];
        outT[((size_t)b * DH + nh * DV + d) * 64 + i] = (_Float16)(o * inv);
    }
}

// ---------------------------------------------------------------------------
// Host side
// ---------------------------------------------------------------------------
static void launch_gemm(const _Float16* Wh, const _Float16* Xh, float* Y, _Float16* Yh,
                        int Cout, int Cin, int Npos, int Bb,
                        const float* g, const float* bb, const float* mm, const float* vv,
                        const float* a, const float* res, const float* mul,
                        const float* rAdd, const float* cAdd, int Wd, hipStream_t s)
{
    dim3 grid((Npos + POSB - 1) / POSB, Cout / MB, Bb);
    wmma_gemm_kernel<<<grid, 256, 0, s>>>(Wh, Xh, Y, Yh, Cout, Cin, Npos,
                                          g, bb, mm, vv, a, res, mul, rAdd, cAdd, Wd);
}

extern "C" void kernel_launch(void* const* d_in, const int* in_sizes, int n_in,
                              void* d_out, int out_size, void* d_ws, size_t ws_size,
                              hipStream_t stream)
{
    (void)in_sizes; (void)n_in; (void)out_size; (void)ws_size;
    const float** in = (const float**)d_in;
    const float* x      = in[0];
    const float* de_w1  = in[1];
    const float *de1_g = in[2],  *de1_b = in[3],  *de1_m = in[4],  *de1_v = in[5];
    const float* de_w2  = in[6];
    const float *de2_g = in[7],  *de2_b = in[8],  *de2_m = in[9],  *de2_v = in[10];
    const float* pp_w1  = in[11];
    const float *pp1_g = in[12], *pp1_b = in[13], *pp1_m = in[14], *pp1_v = in[15];
    const float* pp_a1  = in[16];
    const float* pp_w2  = in[17];
    const float *pp2_g = in[18], *pp2_b = in[19], *pp2_m = in[20], *pp2_v = in[21];
    const float* pp_a2  = in[22];
    const float* wq     = in[23];
    const float* wk     = in[24];
    const float* wv     = in[25];
    const float *pe_rq = in[26], *pe_rk = in[27], *pe_cq = in[28], *pe_ck = in[29];
    const float* per_w  = in[30];
    const float *per_g = in[31], *per_b = in[32], *per_m = in[33], *per_v = in[34];
    const float* per_a  = in[35];
    const float* pec_w  = in[36];
    const float* proj_w = in[37];
    const float *prj_g = in[38], *prj_b = in[39], *prj_m = in[40], *prj_v = in[41];
    const float* proj_a = in[42];
    float* out = (float*)d_out;

    // ---- f32 workspace slots ----
    const size_t SLOT = (size_t)BSZ * DIM * HW;      // 16,777,216
    float* ws   = (float*)d_ws;
    float* s_r   = ws + 0 * SLOT;   // r  -> pool temp
    float* s_qkv = ws + 1 * SLOT;   // qkv_details (live to end)
    float* s_c1  = ws + 2 * SLOT;   // c1
    float* s_p1  = ws + 3 * SLOT;   // p1
    float* s_xp  = ws + 4 * SLOT;   // p2 -> x_pooled
    float* s_p3  = ws + 5 * SLOT;   // p3
    float* sm    = ws + 6 * SLOT;
    const size_t S128 = (size_t)BSZ * NHKD * 64;     // 262144
    const size_t S512 = (size_t)BSZ * DH   * 64;     // 1048576
    float* qmr  = sm;               sm += S128;
    float* kmr  = sm;               sm += S128;
    float* vmr  = sm;               sm += S512;
    float* qmc  = sm;               sm += S128;
    float* kmc  = sm;               sm += S128;
    float* vmc  = sm;               sm += S512;
    float* rowP = sm;               sm += S128;
    float* colP = sm;               sm += S128;

    // ---- f16 region ----
    _Float16* hm = (_Float16*)sm;
    _Float16* xh    = hm;  hm += SLOT;           // x in f16
    _Float16* rh    = hm;  hm += SLOT;           // r in f16 (conv3x3 B)
    _Float16* xph   = hm;  hm += SLOT;           // x_pooled in f16
    _Float16* xprh  = hm;  hm += S128;
    _Float16* xpch  = hm;  hm += S128;
    _Float16* rowTh = hm;  hm += S512;
    _Float16* colTh = hm;  hm += S512;
    _Float16* rowEh = hm;  hm += S512;
    _Float16* colEh = hm;  hm += S512;
    _Float16* wh_de1 = hm; hm += (size_t)DIM * DIM;
    _Float16* wh_pp1 = hm; hm += (size_t)DIM * DIM;
    _Float16* wh_q   = hm; hm += (size_t)DIM * DIM;
    _Float16* wh_k   = hm; hm += (size_t)DIM * DIM;
    _Float16* wh_v   = hm; hm += (size_t)DH * DIM;
    _Float16* wh_per = hm; hm += (size_t)DH * DH;
    _Float16* wh_pec = hm; hm += (size_t)DH * DH;
    _Float16* wh_prj = hm; hm += (size_t)DIM * DH;
    _Float16* wh_PV  = hm; hm += (size_t)DIM * DIM;
    _Float16* wh_w3t = hm; hm += (size_t)9 * DIM * DIM;

    const int NELEM_BLOCKS = (int)(SLOT / 256);

    // 0) one-time f16 conversions (weights + input)
    cvt_h_kernel<<<NELEM_BLOCKS, 256, 0, stream>>>(x, xh);
    cvt_h_kernel<<<(DIM * DIM) / 256, 256, 0, stream>>>(de_w1, wh_de1);
    cvt_h_kernel<<<(DIM * DIM) / 256, 256, 0, stream>>>(pp_w1, wh_pp1);
    cvt_h_kernel<<<(DIM * DIM) / 256, 256, 0, stream>>>(wq, wh_q);
    cvt_h_kernel<<<(DIM * DIM) / 256, 256, 0, stream>>>(wk, wh_k);
    cvt_h_kernel<<<(DH * DIM) / 256, 256, 0, stream>>>(wv, wh_v);
    cvt_h_kernel<<<(DH * DH) / 256, 256, 0, stream>>>(per_w, wh_per);
    cvt_h_kernel<<<(DH * DH) / 256, 256, 0, stream>>>(pec_w, wh_pec);
    cvt_h_kernel<<<(DIM * DH) / 256, 256, 0, stream>>>(proj_w, wh_prj);
    cvt_w3_kernel<<<(9 * DIM * DIM) / 256, 256, 0, stream>>>(de_w2, wh_w3t);

    // 1) r = BN(de_w1 @ x)  (f32 + f16 copies)
    launch_gemm(wh_de1, xh, s_r, rh, DIM, DIM, HW, BSZ,
                de1_g, de1_b, de1_m, de1_v, nullptr, nullptr, nullptr,
                nullptr, nullptr, 1, stream);

    // 2) qkv_details = BN(conv3x3(r)) + r
    {
        dim3 grid(HW / POSB, DIM / MB, BSZ);
        wmma_conv3x3_kernel<<<grid, 256, 0, stream>>>(wh_w3t, rh, s_qkv,
                                                      de2_g, de2_b, de2_m, de2_v, s_r);
    }

    // 3) c1 = PReLU(BN(pp_w1 @ x))
    launch_gemm(wh_pp1, xh, s_c1, nullptr, DIM, DIM, HW, BSZ,
                pp1_g, pp1_b, pp1_m, pp1_v, pp_a1, nullptr, nullptr,
                nullptr, nullptr, 1, stream);

    // 4) p1 = avg3(c1); p2 = avg5(p1); p3 = avg7(p2)   (s_r reused as temp)
    pool_h_kernel<<<NELEM_BLOCKS, 256, 0, stream>>>(s_c1, s_r, 3);
    pool_v_kernel<<<NELEM_BLOCKS, 256, 0, stream>>>(s_r, s_p1, 3);
    pool_h_kernel<<<NELEM_BLOCKS, 256, 0, stream>>>(s_p1, s_r, 5);
    pool_v_kernel<<<NELEM_BLOCKS, 256, 0, stream>>>(s_r, s_xp, 5);   // p2
    pool_h_kernel<<<NELEM_BLOCKS, 256, 0, stream>>>(s_xp, s_r, 7);
    pool_v_kernel<<<NELEM_BLOCKS, 256, 0, stream>>>(s_r, s_p3, 7);   // p3

    // 5) x_pooled = PReLU(BN(dwconv(cat(p1,p1,p3)))) * c1  (f32 + f16)
    dwconv_kernel<<<NELEM_BLOCKS, 256, 0, stream>>>(s_p1, s_p3, pp_w2,
                                                    pp2_g, pp2_b, pp2_m, pp2_v,
                                                    pp_a2, s_c1, s_xp, xph);

    // 6) row/col means of x_pooled (mean commutes with the 1x1 convs)
    mean_kernel<<<(int)(S128 / 256), 256, 0, stream>>>(s_xp, xprh, xpch);

    // 7) q/k/v means via small GEMMs on the means
    launch_gemm(wh_q, xprh, qmr, nullptr, NHKD, DIM, 64, BSZ, nullptr, nullptr, nullptr,
                nullptr, nullptr, nullptr, nullptr, nullptr, nullptr, 1, stream);
    launch_gemm(wh_k, xprh, kmr, nullptr, NHKD, DIM, 64, BSZ, nullptr, nullptr, nullptr,
                nullptr, nullptr, nullptr, nullptr, nullptr, nullptr, 1, stream);
    launch_gemm(wh_v, xprh, vmr, nullptr, DH, DIM, 64, BSZ, nullptr, nullptr, nullptr,
                nullptr, nullptr, nullptr, nullptr, nullptr, nullptr, 1, stream);
    launch_gemm(wh_q, xpch, qmc, nullptr, NHKD, DIM, 64, BSZ, nullptr, nullptr, nullptr,
                nullptr, nullptr, nullptr, nullptr, nullptr, nullptr, 1, stream);
    launch_gemm(wh_k, xpch, kmc, nullptr, NHKD, DIM, 64, BSZ, nullptr, nullptr, nullptr,
                nullptr, nullptr, nullptr, nullptr, nullptr, nullptr, 1, stream);
    launch_gemm(wh_v, xpch, vmc, nullptr, DH, DIM, 64, BSZ, nullptr, nullptr, nullptr,
                nullptr, nullptr, nullptr, nullptr, nullptr, nullptr, 1, stream);

    // 8) axial attention (row uses pe_r*, col uses pe_c*)
    axial_kernel<<<BSZ * NH, 64, 0, stream>>>(qmr, kmr, vmr, pe_rq, pe_rk, rowTh);
    axial_kernel<<<BSZ * NH, 64, 0, stream>>>(qmc, kmc, vmc, pe_cq, pe_ck, colTh);

    // 9) row_e = PReLU(BN(per_w @ row)); col_e = pec_w @ col   (f16 only)
    launch_gemm(wh_per, rowTh, nullptr, rowEh, DH, DH, 64, BSZ,
                per_g, per_b, per_m, per_v, per_a, nullptr, nullptr,
                nullptr, nullptr, 1, stream);
    launch_gemm(wh_pec, colTh, nullptr, colEh, DH, DH, 64, BSZ,
                nullptr, nullptr, nullptr, nullptr, nullptr, nullptr, nullptr,
                nullptr, nullptr, 1, stream);

    // 10) rowP = proj_w @ rowE, colP = proj_w @ colE  (broadcast terms, f32)
    launch_gemm(wh_prj, rowEh, rowP, nullptr, DIM, DH, 64, BSZ,
                nullptr, nullptr, nullptr, nullptr, nullptr, nullptr, nullptr,
                nullptr, nullptr, 1, stream);
    launch_gemm(wh_prj, colEh, colP, nullptr, DIM, DH, 64, BSZ,
                nullptr, nullptr, nullptr, nullptr, nullptr, nullptr, nullptr,
                nullptr, nullptr, 1, stream);

    // 11) composite weight PV = proj_w @ wv   (f16 out, used as GEMM A)
    launch_gemm(wh_prj, wh_v, nullptr, wh_PV, DIM, DH, DIM, 1,
                nullptr, nullptr, nullptr, nullptr, nullptr, nullptr, nullptr,
                nullptr, nullptr, 1, stream);

    // 12) out = PReLU(BN(PV @ x_pooled + rowP + colP)) * qkv_details
    launch_gemm(wh_PV, xph, out, nullptr, DIM, DIM, HW, BSZ,
                prj_g, prj_b, prj_m, prj_v, proj_a, nullptr, s_qkv,
                rowP, colP, WW, stream);
}
